// MultiHeadAttention_30554397344507
// MI455X (gfx1250) — matrix operand order repaired and checked by reading, not compiled, not run on hardware
//
#include <hip/hip_runtime.h>
#include <hip/hip_bf16.h>
#include <math.h>

typedef __attribute__((ext_vector_type(16))) _Float16 v16h;
typedef __attribute__((ext_vector_type(8)))  _Float16 v8h;
typedef __attribute__((ext_vector_type(8)))  float    v8f;
typedef __attribute__((ext_vector_type(4)))  float    v4f;

constexpr int BB = 4;      // batch
constexpr int SS = 1024;   // queries
constexpr int LL = 1024;   // keys
constexpr int DM = 1024;   // d_model
constexpr int NH = 16;     // heads
constexpr int DH = 64;     // head dim

static __device__ __forceinline__ v8f wmma_f16(v16h a, v16h b, v8f c) {
  // D = A(16x32 f16) * B(32x16 f16) + C(16x16 f32)
  return __builtin_amdgcn_wmma_f32_16x16x32_f16(false, a, false, b, (short)0, c,
                                                false, false);
}

static __device__ __forceinline__ float redmax16(float x) {
#pragma unroll
  for (int m = 8; m >= 1; m >>= 1) x = fmaxf(x, __shfl_xor(x, m, 32));
  return x;
}
static __device__ __forceinline__ float redsum16(float x) {
#pragma unroll
  for (int m = 8; m >= 1; m >>= 1) x += __shfl_xor(x, m, 32);
  return x;
}

// ---------------------------------------------------------------- stage 0 --
// Convert Wq,Wk,Wv (f32, [1024x1024] each, row-major out x in) to f16 once.
__global__ void cvt_w(const float* __restrict__ Wq, const float* __restrict__ Wk,
                      const float* __restrict__ Wv, _Float16* __restrict__ Wh) {
  size_t i = (size_t)blockIdx.x * blockDim.x + threadIdx.x;
  if (i >= (size_t)3 * DM * DM) return;
  size_t w = i >> 20, j = i & ((1u << 20) - 1);
  float x = (w == 0) ? Wq[j] : (w == 1) ? Wk[j] : Wv[j];
  Wh[i] = (_Float16)x;
}

// ---------------------------------------------------------------- stage 1 --
// Y[m,n] = sum_k X[m,k] * W[n,k]   (torch Linear: x @ W.T)
// One wave computes a 64(M) x 64(N) tile: B-fragments loaded once per k-step
// and reused across 4 row tiles (16 WMMAs per 32-wide k-step).
// Q,K -> [B,H,T,64]; V -> [B,H,64,L] (transposed for PV B-fragments).
__global__ void __launch_bounds__(128) mha_proj(
    const float* __restrict__ q, const float* __restrict__ k,
    const float* __restrict__ v, const _Float16* __restrict__ Wh,
    _Float16* __restrict__ Qh, _Float16* __restrict__ Kh,
    _Float16* __restrict__ Vt) {
  const int lane = threadIdx.x & 31;
  const int wave = threadIdx.x >> 5;
  const int g = lane >> 4;   // lane group (0: lanes 0-15, 1: lanes 16-31)
  const int ln = lane & 15;

  const int task = blockIdx.x * 4 + wave;  // 0 .. 3*1024-1
  const int proj = task >> 10;             // 0=Q 1=K 2=V
  const int t = task & 1023;
  const int mt = t >> 4;                   // 0..63   (64-row block of 4096 rows)
  const int h = t & 15;                    // head == 64-wide N strip
  const int m0 = mt * 64;

  const float* X = (proj == 0) ? q : (proj == 1) ? k : v;
  const _Float16* W = Wh + (size_t)proj * DM * DM;

  v8f acc[4][4];                           // [row tile][col tile]
#pragma unroll
  for (int mi = 0; mi < 4; ++mi)
#pragma unroll
    for (int t2 = 0; t2 < 4; ++t2)
#pragma unroll
      for (int r = 0; r < 8; ++r) acc[mi][t2][r] = 0.f;

  for (int kk = 0; kk < DM; kk += 32) {
    // B fragments: element (K=kk+16g+e, N=16*t2+ln) = W[n][k], 16 contiguous f16
    v16h Bf[4];
#pragma unroll
    for (int t2 = 0; t2 < 4; ++t2)
      Bf[t2] = *(const v16h*)(W + (size_t)(h * DH + 16 * t2 + ln) * DM + kk + 16 * g);

#pragma unroll
    for (int mi = 0; mi < 4; ++mi) {
      // A fragment: lane holds contiguous K run [16g..16g+15] (matching B)
      // for row m=lane%16; convert f32 -> f16 in flight.
      const float* xa = X + (size_t)(m0 + mi * 16 + ln) * DM + kk + 16 * g;
      const float* xb = xa + 8;
      v4f a0 = ((const v4f*)xa)[0], a1 = ((const v4f*)xa)[1];
      v4f a2 = ((const v4f*)xb)[0], a3 = ((const v4f*)xb)[1];
      v16h A;
#pragma unroll
      for (int e = 0; e < 4; ++e) {
        A[e] = (_Float16)a0[e];
        A[e + 4] = (_Float16)a1[e];
        A[e + 8] = (_Float16)a2[e];
        A[e + 12] = (_Float16)a3[e];
      }
#pragma unroll
      for (int t2 = 0; t2 < 4; ++t2)
        acc[mi][t2] = wmma_f16(A, Bf[t2], acc[mi][t2]);
    }
  }

  const int b = m0 >> 10;        // 1024 rows per batch; 64-row blocks never straddle
  const int s0 = m0 & 1023;
  const int bh = b * NH + h;
#pragma unroll
  for (int mi = 0; mi < 4; ++mi)
#pragma unroll
    for (int t2 = 0; t2 < 4; ++t2)
#pragma unroll
      for (int r = 0; r < 8; ++r) {
        _Float16 val = (_Float16)acc[mi][t2][r];
        const int srow = s0 + mi * 16 + r + 8 * g;   // D: M = r + 8g
        const int dcol = 16 * t2 + ln;               // D: N = lane%16
        if (proj == 0)       Qh[((size_t)bh * SS + srow) * DH + dcol] = val;
        else if (proj == 1)  Kh[((size_t)bh * LL + srow) * DH + dcol] = val;
        else                 Vt[((size_t)bh * DH + dcol) * LL + srow] = val;
      }
}

// ---------------------------------------------------------------- stage 2 --
// Flash attention. Block = 4 waves = 128 queries of one (b,h); wave = 32
// queries (2 row tiles) so every K/V fragment feeds 2 WMMAs.
__global__ void __launch_bounds__(128) mha_attn(
    const _Float16* __restrict__ Qh, const _Float16* __restrict__ Kh,
    const _Float16* __restrict__ Vt, const unsigned char* __restrict__ mask,
    float* __restrict__ out) {
  const int lane = threadIdx.x & 31;
  const int wave = threadIdx.x >> 5;
  const int g = lane >> 4;
  const int ln = lane & 15;

  const int qblk = blockIdx.x & 7;    // S/128 = 8
  const int bh = blockIdx.x >> 3;
  const int b = bh >> 4;              // NH = 16
  const int h = bh & 15;
  const int q0 = qblk * 128 + wave * 32;

  const _Float16* Qbase = Qh + ((size_t)bh * SS + q0) * DH;
  const _Float16* Kbase = Kh + (size_t)bh * LL * DH;
  const _Float16* Vbase = Vt + (size_t)bh * DH * LL;
  const unsigned char* mk = mask + (size_t)b * LL;

  __shared__ _Float16 Plds[4][2][16 * 32];  // per-wave P scratch (C->A transpose)

  // Q A-fragments: 2 row tiles x 2 dh chunks of K=32
  v16h qA[2][2];
#pragma unroll
  for (int mi = 0; mi < 2; ++mi) {
    const _Float16* qrow = Qbase + (mi * 16 + ln) * DH;
#pragma unroll
    for (int c = 0; c < 2; ++c) {
      v8h lo = *(const v8h*)(qrow + 32 * c + 16 * g);
      v8h hi = *(const v8h*)(qrow + 32 * c + 16 * g + 8);
      qA[mi][c] = __builtin_shufflevector(lo, hi, 0, 1, 2, 3, 4, 5, 6, 7, 8, 9,
                                          10, 11, 12, 13, 14, 15);
    }
  }

  float mrow[2][8], lrow[2][8];
  v8f o[2][4];
#pragma unroll
  for (int mi = 0; mi < 2; ++mi) {
#pragma unroll
    for (int r = 0; r < 8; ++r) { mrow[mi][r] = -INFINITY; lrow[mi][r] = 0.f; }
#pragma unroll
    for (int t = 0; t < 4; ++t)
#pragma unroll
      for (int r = 0; r < 8; ++r) o[mi][t][r] = 0.f;
  }

  const float scale = 0.125f;  // 1/sqrt(64)

  for (int kb = 0; kb < LL; kb += 32) {
    // K B-fragments: element (K=dh 32c+16g+e, N=key kb+16t+ln) = K[key][dh]
    const _Float16* k0 = Kbase + (size_t)(kb + ln) * DH + 16 * g;
    const _Float16* k1 = k0 + 16 * DH;
    v16h bk00 = *(const v16h*)(k0);
    v16h bk01 = *(const v16h*)(k0 + 32);
    v16h bk10 = *(const v16h*)(k1);
    v16h bk11 = *(const v16h*)(k1 + 32);

    if (kb + 32 < LL) {  // pull next block toward the WGP while we compute
      __builtin_prefetch(Kbase + (size_t)(kb + 32 + ln) * DH, 0, 1);
      __builtin_prefetch(Vbase + (size_t)ln * LL + kb + 32, 0, 1);
    }

    const bool pm0 = mk[kb + ln] != 0;
    const bool pm1 = mk[kb + 16 + ln] != 0;

#pragma unroll
    for (int mi = 0; mi < 2; ++mi) {
      v8f c0, c1;
#pragma unroll
      for (int r = 0; r < 8; ++r) { c0[r] = 0.f; c1[r] = 0.f; }
      c0 = wmma_f16(qA[mi][0], bk00, c0);
      c0 = wmma_f16(qA[mi][1], bk01, c0);
      c1 = wmma_f16(qA[mi][0], bk10, c1);
      c1 = wmma_f16(qA[mi][1], bk11, c1);

      _Float16* P = Plds[wave][mi];
      // ---- online softmax per row (row lives in one VGPR across 16 lanes)
#pragma unroll
      for (int r = 0; r < 8; ++r) {
        float s0 = pm0 ? -1e9f : c0[r] * scale;   // masked_fill then softmax
        float s1 = pm1 ? -1e9f : c1[r] * scale;
        float mx = redmax16(fmaxf(s0, s1));
        float mn = fmaxf(mrow[mi][r], mx);
        float alpha = __expf(mrow[mi][r] - mn);
        mrow[mi][r] = mn;
        float p0 = __expf(s0 - mn);
        float p1 = __expf(s1 - mn);
        lrow[mi][r] = lrow[mi][r] * alpha + redsum16(p0 + p1);
#pragma unroll
        for (int t = 0; t < 4; ++t) o[mi][t][r] *= alpha;
        // stash P in LDS: C layout (M=r+8g, N=ln) -> row-major [16 x 32]
        P[(r + 8 * g) * 32 + ln] = (_Float16)p0;
        P[(r + 8 * g) * 32 + 16 + ln] = (_Float16)p1;
      }
    }

    // DS ops within a wave are in-order; drain + stop compiler reordering
    asm volatile("s_wait_dscnt 0x0" ::: "memory");

    // re-load P as A fragments (row = lane, contiguous K run 16g..16g+15)
    v16h pA[2];
#pragma unroll
    for (int mi = 0; mi < 2; ++mi) {
      const _Float16* P = Plds[wave][mi];
      v8h plo = *(const v8h*)(P + ln * 32 + 16 * g);
      v8h phi = *(const v8h*)(P + ln * 32 + 16 * g + 8);
      pA[mi] = __builtin_shufflevector(plo, phi, 0, 1, 2, 3, 4, 5, 6, 7, 8, 9,
                                       10, 11, 12, 13, 14, 15);
    }

    // ---- O += P * V; V transposed so B frag (K=key, N=dh) is contiguous in K,
    // and each V fragment feeds both row tiles.
#pragma unroll
    for (int t = 0; t < 4; ++t) {
      v16h bv = *(const v16h*)(Vbase + (size_t)(16 * t + ln) * LL + kb + 16 * g);
#pragma unroll
      for (int mi = 0; mi < 2; ++mi) o[mi][t] = wmma_f16(pA[mi], bv, o[mi][t]);
    }
  }

  // ---- epilogue: O / l, write [B,S,1024] f32, coalesced across lanes
  float* obase = out + ((size_t)b * SS + q0) * DM + h * DH;
#pragma unroll
  for (int mi = 0; mi < 2; ++mi)
#pragma unroll
    for (int r = 0; r < 8; ++r) {
      float inv = 1.0f / lrow[mi][r];
      float* orow = obase + (size_t)(mi * 16 + r + 8 * g) * DM;
#pragma unroll
      for (int t = 0; t < 4; ++t) orow[16 * t + ln] = o[mi][t][r] * inv;
    }
}

// -------------------------------------------------------------------------
extern "C" void kernel_launch(void* const* d_in, const int* in_sizes, int n_in,
                              void* d_out, int out_size, void* d_ws,
                              size_t ws_size, hipStream_t stream) {
  (void)in_sizes; (void)n_in; (void)out_size; (void)ws_size;
  const float* q = (const float*)d_in[0];
  const float* k = (const float*)d_in[1];
  const float* v = (const float*)d_in[2];
  const float* Wq = (const float*)d_in[3];
  const float* Wk = (const float*)d_in[4];
  const float* Wv = (const float*)d_in[5];
  const unsigned char* mask = (const unsigned char*)d_in[6];  // jnp.bool_ = 1 byte
  float* out = (float*)d_out;

  // workspace layout (f16 elements): W(3M) | Q(4M) | K(4M) | Vt(4M) = 31.5 MB
  _Float16* ws = (_Float16*)d_ws;
  _Float16* Wh = ws;
  _Float16* Qh = Wh + (size_t)3 * DM * DM;
  _Float16* Kh = Qh + (size_t)BB * NH * SS * DH;
  _Float16* Vt = Kh + (size_t)BB * NH * LL * DH;

  cvt_w<<<(3 * DM * DM + 255) / 256, 256, 0, stream>>>(Wq, Wk, Wv, Wh);
  // 3 projections x 64 M-blocks x 16 N-strips = 3072 wave-tasks / 4 waves
  mha_proj<<<768, 128, 0, stream>>>(q, k, v, Wh, Qh, Kh, Vt);
  // B*H*(S/128) = 512 blocks, 4 waves each
  mha_attn<<<BB * NH * (SS / 128), 128, 0, stream>>>(Qh, Kh, Vt, mask, out);
}
